// GINNet_45019847197002
// MI455X (gfx1250) — matrix operand order repaired and checked by reading, not compiled, not run on hardware
//
#include <hip/hip_runtime.h>
#include <hip/hip_bf16.h>

// ---------------- problem constants ----------------
#define NN   100000     // nodes
#define NE   3200000    // edges
#define NG   2048       // graphs
#define MD   78         // input feature dim
#define KPAD 96         // layer-0 K padded to 3x32
#define HID  32
#define OUTD 128
#define BN_EPS 1e-5f

typedef __attribute__((ext_vector_type(16))) __bf16 v16bf;
typedef __attribute__((ext_vector_type(8)))  float  v8f;

union U16 { v16bf v; __bf16 e[16]; };
union U8  { v8f  v; float  e[8];  };

// ---------------------------------------------------------------------------
// Tensor Data Mover: load a contiguous [16 x rowElems] f32 tile into LDS with
// row padding (interval 32 dwords, pad 4 dwords -> LDS row stride 36*rowElems/32 dwords,
// 16B-aligned rows, bank-spread for the transposed reads).
// ---------------------------------------------------------------------------
#if __has_builtin(__builtin_amdgcn_tensor_load_to_lds) && __has_builtin(__builtin_amdgcn_s_wait_tensorcnt)
#define CDNA5_TDM 1
typedef __attribute__((ext_vector_type(4))) unsigned int v4u;
typedef __attribute__((ext_vector_type(4))) int v4i;
typedef __attribute__((ext_vector_type(8))) int v8i;

__device__ __forceinline__ void tdm_load_tile(unsigned ldsOff, const void* gptr, int rowElems)
{
    unsigned long long ga = (unsigned long long)(size_t)gptr;
    unsigned galo = (unsigned)__builtin_amdgcn_readfirstlane((int)(unsigned)ga);
    unsigned gahi = (unsigned)__builtin_amdgcn_readfirstlane((int)(unsigned)(ga >> 32));
    unsigned lof  = (unsigned)__builtin_amdgcn_readfirstlane((int)ldsOff);
    unsigned re   = (unsigned)__builtin_amdgcn_readfirstlane(rowElems);

    // D# group 0: count=1 | lds_addr | global_addr | type=2
    v4u g0;
    g0.x = 1u;
    g0.y = lof;
    g0.z = galo;
    g0.w = (gahi & 0x01FFFFFFu) | (2u << 30);

    // D# group 1: data_size=4B, pad_enable, pad_interval=32dw, pad_amount=4dw
    // tensor_dim0 = rowElems, tensor_dim1 = 16, tile_dim0 = rowElems,
    // tile_dim1 = 16, tile_dim2 = 0, tensor_dim0_stride = rowElems
    v8i g1;
    g1[0] = (int)((2u << 16) | (1u << 20) | (4u << 22) | (3u << 25));
    g1[1] = (int)(re << 16);                 // tensor_dim0[15:0] @ bits[31:16]
    g1[2] = (int)((re >> 16) | (16u << 16)); // tensor_dim0[31:16] | tensor_dim1[15:0]
    g1[3] = (int)(re << 16);                 // tensor_dim1[31:16]=0 | tile_dim0
    g1[4] = 16;                              // tile_dim1 | tile_dim2=0
    g1[5] = (int)re;                         // tensor_dim0_stride[31:0]
    g1[6] = 0;
    g1[7] = 0;

    v4i gz = {0, 0, 0, 0};
#if defined(__clang_major__) && __clang_major__ >= 23
    v8i gz8 = {0, 0, 0, 0, 0, 0, 0, 0};
    __builtin_amdgcn_tensor_load_to_lds(g0, g1, gz, gz, gz8, 0);
#else
    __builtin_amdgcn_tensor_load_to_lds(g0, g1, gz, gz, 0);
#endif
}
#endif

// ---------------------------------------------------------------------------
// Pack a row-major weight W[Kreal x N] (f32) into WMMA-B bf16 lane layout
// (zero-padded past Kreal): one contiguous 32B vector per lane per tile.
// ---------------------------------------------------------------------------
__global__ __launch_bounds__(256) void pack_b(const float* __restrict__ W,
                                              int Kreal, int N, int Kchunks,
                                              __bf16* __restrict__ out)
{
    int ntiles = N >> 4;
    int total  = Kchunks * ntiles * 32 * 16;
    int idx = blockIdx.x * blockDim.x + threadIdx.x;
    if (idx >= total) return;
    int e    = idx & 15;
    int lane = (idx >> 4) & 31;
    int rest = idx >> 9;
    int t    = rest % ntiles;
    int kc   = rest / ntiles;
    int k = kc * 32 + ((lane & 16) ? 8 : 0) + (e < 8 ? e : e + 8);
    float v = (k < Kreal) ? W[(size_t)k * N + t * 16 + (lane & 15)] : 0.f;
    out[idx] = (__bf16)v;
}

// ---------------------------------------------------------------------------
// Layer 0: agg[n][0..77] = x[n][0..77], agg[n][78..95] = 0  (K padded to 96)
// ---------------------------------------------------------------------------
__global__ __launch_bounds__(256) void init_agg0(const float* __restrict__ x,
                                                 float* __restrict__ agg)
{
    int i = blockIdx.x * blockDim.x + threadIdx.x;
    if (i >= NN * KPAD) return;
    int n = i / KPAD, c = i - n * KPAD;
    agg[i] = (c < MD) ? x[(size_t)n * MD + c] : 0.f;
}

// ---------------------------------------------------------------------------
// Layers 1..4: agg[n*32+c] = scale[c]*h[n*32+c] + shift[c]  (prev BN fused)
// ---------------------------------------------------------------------------
__global__ __launch_bounds__(256) void init_agg(const float* __restrict__ h,
                                                const float* __restrict__ ss,
                                                float* __restrict__ agg, int n)
{
    int i = blockIdx.x * blockDim.x + threadIdx.x;
    if (i >= n) return;
    int c = i & 31;
    agg[i] = ss[c] * h[i] + ss[32 + c];
}

// ---------------------------------------------------------------------------
// One wave per edge; lanes sweep channels. agg[dst] += bn(h[src]).
// global_atomic_add_f32; agg is L2-resident.
// ---------------------------------------------------------------------------
__global__ __launch_bounds__(256) void edge_scatter(const int* __restrict__ src,
                                                    const int* __restrict__ dst,
                                                    const float* __restrict__ h, int ldh,
                                                    const float* __restrict__ ss,
                                                    float* __restrict__ agg, int ldagg,
                                                    int nEdges)
{
    int wid  = (blockIdx.x * blockDim.x + threadIdx.x) >> 5;
    int lane = threadIdx.x & 31;
    if (wid >= nEdges) return;
    int s = src[wid], d = dst[wid];
    const float* hs = h   + (size_t)s * ldh;
    float*       ad = agg + (size_t)d * ldagg;
    for (int c = lane; c < ldh; c += 32) {
        float v = hs[c];
        if (ss) v = ss[c] * v + ss[32 + c];
        unsafeAtomicAdd(ad + c, v);
    }
}

// ---------------------------------------------------------------------------
// Fused GIN node MLP: h = ReLU( ReLU(agg@W1+b1) @ W2 + b2 ) + BN stats.
// One wave per 16-row M-tile. GEMM1 A tile staged in LDS by the TDM
// (fallback: direct float4 global loads). LDS A region is reused as the
// GEMM1->GEMM2 transpose tile once A is consumed.
// ---------------------------------------------------------------------------
__global__ __launch_bounds__(256) void mlp_wmma(const float* __restrict__ agg,
                                                int Kchunks,
                                                const __bf16* __restrict__ W1p,
                                                const float* __restrict__ b1,
                                                const __bf16* __restrict__ W2p,
                                                const float* __restrict__ b2,
                                                float* __restrict__ hout,
                                                float* __restrict__ stats,
                                                int nNodes)
{
    __shared__ float ldsbuf[8][16 * 108];       // 54KB: max A tile (16x96 + pads)
    const int wave = threadIdx.x >> 5;
    const int lane = threadIdx.x & 31;
    const int half = lane >> 4;                 // 0: k 0-7,16-23; 1: k 8-15,24-31
    const int l15  = lane & 15;
    const int mtile = blockIdx.x * 8 + wave;
    const int row0  = mtile * 16;
    const bool active = (row0 + 15) < nNodes;   // NN % 16 == 0 -> full tiles
    const int row0c = active ? row0 : 0;        // clamp tail waves (no branches)
    const int kb  = half ? 8 : 0;
    const int ldA = Kchunks * 32;

#if defined(CDNA5_TDM)
    tdm_load_tile((unsigned)(size_t)&ldsbuf[wave][0],
                  agg + (size_t)row0c * ldA, ldA);
#endif

    // ---------------- GEMM1: agg @ W1 ----------------
#if defined(CDNA5_TDM)
    __builtin_amdgcn_s_wait_tensorcnt(0);
    __asm volatile("" ::: "memory");
    const float* rp = &ldsbuf[wave][0] + l15 * (Kchunks * 36);  // padded row stride
    const int cstep = 36;
#else
    const float* rp = agg + (size_t)(row0c + l15) * ldA;
    const int cstep = 32;
#endif
    v8f acc[2] = {{}, {}};
    for (int kc = 0; kc < Kchunks; ++kc) {
        const float* cp = rp + kc * cstep + kb;
        float4 f0 = *(const float4*)(cp);
        float4 f1 = *(const float4*)(cp + 4);
        float4 f2 = *(const float4*)(cp + 16);
        float4 f3 = *(const float4*)(cp + 20);
        U16 A;
        A.e[0]  = (__bf16)f0.x; A.e[1]  = (__bf16)f0.y; A.e[2]  = (__bf16)f0.z; A.e[3]  = (__bf16)f0.w;
        A.e[4]  = (__bf16)f1.x; A.e[5]  = (__bf16)f1.y; A.e[6]  = (__bf16)f1.z; A.e[7]  = (__bf16)f1.w;
        A.e[8]  = (__bf16)f2.x; A.e[9]  = (__bf16)f2.y; A.e[10] = (__bf16)f2.z; A.e[11] = (__bf16)f2.w;
        A.e[12] = (__bf16)f3.x; A.e[13] = (__bf16)f3.y; A.e[14] = (__bf16)f3.z; A.e[15] = (__bf16)f3.w;
#pragma unroll
        for (int t = 0; t < 2; ++t) {
            v16bf B = *(const v16bf*)(W1p + ((size_t)(kc * 2 + t) * 32 + lane) * 16);
            acc[t] = __builtin_amdgcn_wmma_f32_16x16x32_bf16(
                false, A.v, false, B, (short)0, acc[t], false, false);
        }
    }

    // bias + ReLU -> transpose tile in LDS (reuse A region; A fully consumed)
    float* T = &ldsbuf[wave][0];                // viewed as [16][33]
#pragma unroll
    for (int t = 0; t < 2; ++t) {
        U8 C; C.v = acc[t];
        float bb = b1[t * 16 + l15];
#pragma unroll
        for (int r = 0; r < 8; ++r) {
            T[(r + half * 8) * 33 + t * 16 + l15] = fmaxf(C.e[r] + bb, 0.f);
        }
    }
    __syncthreads();

    // ---------------- GEMM2: h1 @ W2 ----------------
    U16 A2;
    const float* tr = T + l15 * 33 + kb;
#pragma unroll
    for (int j = 0; j < 8; ++j) {
        A2.e[j]     = (__bf16)tr[j];
        A2.e[8 + j] = (__bf16)tr[16 + j];
    }
    v8f acc2[2] = {{}, {}};
#pragma unroll
    for (int t = 0; t < 2; ++t) {
        v16bf B = *(const v16bf*)(W2p + ((size_t)t * 32 + lane) * 16);
        acc2[t] = __builtin_amdgcn_wmma_f32_16x16x32_bf16(
            false, A2.v, false, B, (short)0, acc2[t], false, false);
    }

    // bias + ReLU, store h, accumulate BN stats
#pragma unroll
    for (int t = 0; t < 2; ++t) {
        U8 C; C.v = acc2[t];
        int n = t * 16 + l15;
        float bb = b2[n];
        float s = 0.f, q = 0.f;
#pragma unroll
        for (int r = 0; r < 8; ++r) {
            float v = fmaxf(C.e[r] + bb, 0.f);
            s += v; q += v * v;
            if (active) hout[(size_t)(row0 + r + half * 8) * 32 + n] = v;
        }
        s += __shfl_xor(s, 16, 32);             // combine the two M-halves
        q += __shfl_xor(q, 16, 32);
        if (active && half == 0) {
            unsafeAtomicAdd(stats + n, s);
            unsafeAtomicAdd(stats + 32 + n, q);
        }
    }
}

// ---------------------------------------------------------------------------
// stats -> (scale, shift):  scale=g*rsqrt(var+eps); shift=beta-mean*scale
// ---------------------------------------------------------------------------
__global__ void bn_finalize(const float* __restrict__ stats,
                            const float* __restrict__ gamma,
                            const float* __restrict__ beta,
                            float* __restrict__ ss, float invN)
{
    int c = threadIdx.x;                        // 32 threads
    float mean  = stats[c] * invN;
    float var   = stats[32 + c] * invN - mean * mean;
    float scale = gamma[c] * rsqrtf(var + BN_EPS);
    ss[c]      = scale;
    ss[32 + c] = beta[c] - mean * scale;
}

// ---------------------------------------------------------------------------
// pooled[g] += bn(h[n]) for sorted batch; one wave per 32 consecutive rows,
// register-accumulate per segment, one atomic per (graph, channel) segment.
// ---------------------------------------------------------------------------
__global__ __launch_bounds__(256) void pool_bn(const float* __restrict__ h,
                                               const float* __restrict__ ss,
                                               const int* __restrict__ batch,
                                               float* __restrict__ pooled, int nNodes)
{
    int wid  = (blockIdx.x * blockDim.x + threadIdx.x) >> 5;
    int lane = threadIdx.x & 31;
    int r0   = wid * 32;
    if (r0 >= nNodes) return;
    int rend = min(r0 + 32, nNodes);
    float sc = ss[lane], sh = ss[32 + lane];
    float acc = 0.f;
    int g = batch[r0];
    for (int r = r0; r < rend; ++r) {
        int gr = batch[r];
        if (gr != g) {
            unsafeAtomicAdd(pooled + (size_t)g * 32 + lane, acc);
            acc = 0.f; g = gr;
        }
        acc += sc * h[(size_t)r * 32 + lane] + sh;
    }
    unsafeAtomicAdd(pooled + (size_t)g * 32 + lane, acc);
}

// ---------------------------------------------------------------------------
// out = ReLU(pooled @ fc_w + fc_b): block = M-tile (16 rows), wave = N-tile.
// ---------------------------------------------------------------------------
__global__ __launch_bounds__(256) void fc_wmma(const float* __restrict__ pooled,
                                               const __bf16* __restrict__ Wp,
                                               const float* __restrict__ bias,
                                               float* __restrict__ out)
{
    int wave = threadIdx.x >> 5;                // N-tile 0..7
    int lane = threadIdx.x & 31;
    int half = lane >> 4, l15 = lane & 15;
    int mtile = blockIdx.x;                     // 0..127  (NG/16)
    int kb  = half ? 8 : 0;

    const float* ap = pooled + (size_t)(mtile * 16 + l15) * 32 + kb;
    float4 f0 = *(const float4*)(ap);
    float4 f1 = *(const float4*)(ap + 4);
    float4 f2 = *(const float4*)(ap + 16);
    float4 f3 = *(const float4*)(ap + 20);
    U16 A;
    A.e[0]  = (__bf16)f0.x; A.e[1]  = (__bf16)f0.y; A.e[2]  = (__bf16)f0.z; A.e[3]  = (__bf16)f0.w;
    A.e[4]  = (__bf16)f1.x; A.e[5]  = (__bf16)f1.y; A.e[6]  = (__bf16)f1.z; A.e[7]  = (__bf16)f1.w;
    A.e[8]  = (__bf16)f2.x; A.e[9]  = (__bf16)f2.y; A.e[10] = (__bf16)f2.z; A.e[11] = (__bf16)f2.w;
    A.e[12] = (__bf16)f3.x; A.e[13] = (__bf16)f3.y; A.e[14] = (__bf16)f3.z; A.e[15] = (__bf16)f3.w;

    v16bf B = *(const v16bf*)(Wp + ((size_t)wave * 32 + lane) * 16);
    v8f acc = {};
    acc = __builtin_amdgcn_wmma_f32_16x16x32_bf16(
        false, A.v, false, B, (short)0, acc, false, false);

    U8 C; C.v = acc;
    int n = wave * 16 + l15;
    float bb = bias[n];
#pragma unroll
    for (int r = 0; r < 8; ++r) {
        int m = mtile * 16 + r + half * 8;
        out[(size_t)m * OUTD + n] = fmaxf(C.e[r] + bb, 0.f);
    }
}

// ---------------------------------------------------------------------------
extern "C" void kernel_launch(void* const* d_in, const int* in_sizes, int n_in,
                              void* d_out, int out_size, void* d_ws, size_t ws_size,
                              hipStream_t stream)
{
    (void)in_sizes; (void)n_in; (void)out_size; (void)ws_size;

    const float* x     = (const float*)d_in[0];
    const int*   ei    = (const int*)  d_in[1];
    const int*   batch = (const int*)  d_in[2];
    const float* w1_0  = (const float*)d_in[3];
    const float* b1_0  = (const float*)d_in[4];
    const float* w2_0  = (const float*)d_in[5];
    const float* b2_0  = (const float*)d_in[6];
    const float* w1    = (const float*)d_in[7];
    const float* b1s   = (const float*)d_in[8];
    const float* w2    = (const float*)d_in[9];
    const float* b2s   = (const float*)d_in[10];
    const float* gamma = (const float*)d_in[11];
    const float* beta  = (const float*)d_in[12];
    const float* fc_w  = (const float*)d_in[13];
    const float* fc_b  = (const float*)d_in[14];
    float* out = (float*)d_out;

    const int* src = ei;
    const int* dst = ei + NE;

    // ---- carve workspace (256B aligned) ----
    char* ws = (char*)d_ws;
    size_t off = 0;
    auto carve = [&](size_t bytes) -> void* {
        void* p = ws + off;
        off += (bytes + 255) & ~(size_t)255;
        return p;
    };
    float*  agg    = (float*) carve((size_t)NN * KPAD * 4);  // 38.4 MB (reused NNx32)
    float*  hA     = (float*) carve((size_t)NN * HID * 4);   // 12.8 MB
    float*  hB     = (float*) carve((size_t)NN * HID * 4);   // 12.8 MB
    float*  pooled = (float*) carve((size_t)NG * HID * 4);   // 256 KB
    float*  stats  = (float*) carve(5 * 64 * 4);
    float*  ssbuf  = (float*) carve(5 * 64 * 4);
    __bf16* w10p   = (__bf16*)carve(3 * 2 * 512 * 2);        // K padded to 96
    __bf16* w20p   = (__bf16*)carve(2 * 512 * 2);
    __bf16* w1p    = (__bf16*)carve(4 * 2 * 512 * 2);
    __bf16* w2p    = (__bf16*)carve(4 * 2 * 512 * 2);
    __bf16* fcp    = (__bf16*)carve(8 * 512 * 2);

    // ---- pack weights into WMMA-B layout ----
    pack_b<<<(3 * 2 * 512 + 255) / 256, 256, 0, stream>>>(w1_0, MD, HID, 3, w10p);
    pack_b<<<4, 256, 0, stream>>>(w2_0, HID, HID, 1, w20p);
    for (int i = 0; i < 4; ++i) {
        pack_b<<<4, 256, 0, stream>>>(w1 + (size_t)i * HID * HID, HID, HID, 1, w1p + (size_t)i * 1024);
        pack_b<<<4, 256, 0, stream>>>(w2 + (size_t)i * HID * HID, HID, HID, 1, w2p + (size_t)i * 1024);
    }
    pack_b<<<16, 256, 0, stream>>>(fc_w, HID, OUTD, 1, fcp);

    hipMemsetAsync(stats, 0, 5 * 64 * 4, stream);
    hipMemsetAsync(pooled, 0, (size_t)NG * HID * 4, stream);

    const float invN = 1.0f / (float)NN;
    const int edgeBlocks = (int)(((size_t)NE * 32 + 255) / 256);   // 400000
    const int mlpBlocks  = (NN / 16 + 7) / 8;                      // 782

    // ---- layer 0: agg = x (padded) + scatter(x); MLP(96->32->32) ----
    init_agg0<<<(NN * KPAD + 255) / 256, 256, 0, stream>>>(x, agg);
    edge_scatter<<<edgeBlocks, 256, 0, stream>>>(src, dst, x, MD, nullptr, agg, KPAD, NE);
    mlp_wmma<<<mlpBlocks, 256, 0, stream>>>(agg, 3, w10p, b1_0, w20p, b2_0,
                                            hA, stats, NN);
    bn_finalize<<<1, 32, 0, stream>>>(stats, gamma, beta, ssbuf, invN);

    // ---- layers 1..4 (BN of previous layer fused into aggregation) ----
    for (int i = 1; i < 5; ++i) {
        const float* hin  = ((i - 1) & 1) ? hB : hA;   // 1:hA 2:hB 3:hA 4:hB
        float*       hout = ((i - 1) & 1) ? hA : hB;
        const float* ssp  = ssbuf + (size_t)(i - 1) * 64;
        init_agg<<<(NN * HID + 255) / 256, 256, 0, stream>>>(hin, ssp, agg, NN * HID);
        edge_scatter<<<edgeBlocks, 256, 0, stream>>>(src, dst, hin, HID, ssp, agg, HID, NE);
        mlp_wmma<<<mlpBlocks, 256, 0, stream>>>(agg, 1,
                                                w1p + (size_t)(i - 1) * 1024, b1s + (size_t)(i - 1) * HID,
                                                w2p + (size_t)(i - 1) * 1024, b2s + (size_t)(i - 1) * HID,
                                                hout, stats + (size_t)i * 64, NN);
        bn_finalize<<<1, 32, 0, stream>>>(stats + (size_t)i * 64,
                                          gamma + (size_t)i * HID, beta + (size_t)i * HID,
                                          ssbuf + (size_t)i * 64, invN);
    }

    // ---- pool (BN of layer 4 fused) + fc ----
    const int poolBlocks = (int)((((size_t)(NN + 31) / 32) * 32 + 255) / 256);
    pool_bn<<<poolBlocks, 256, 0, stream>>>(hA, ssbuf + 4 * 64, batch, pooled, NN);
    fc_wmma<<<NG / 16, 256, 0, stream>>>(pooled, fcp, fc_b, out);
}
